// MultiHeadAttention_13280038879603
// MI455X (gfx1250) — compile-verified, matrix operand order checked by reading, not run on hardware
//
#include <hip/hip_runtime.h>
#include <hip/hip_bf16.h>

typedef __attribute__((ext_vector_type(16))) _Float16 v16h;
typedef __attribute__((ext_vector_type(8)))  _Float16 v8h;
typedef __attribute__((ext_vector_type(8)))  float    v8f;

typedef __attribute__((ext_vector_type(4))) unsigned tdm_g0_t;
typedef __attribute__((ext_vector_type(8))) int      tdm_g1_t;
typedef __attribute__((ext_vector_type(4))) int      tdm_g2_t;

// Problem constants (from reference): B=2, N=2048, C=1024, H=16, D=64
#define BB 2
#define NN 2048
#define CC 1024
#define HH 16
#define DD 64
#define TOK (BB * NN)      // 4096 tokens
#define JQKV (3 * CC)      // 3072 qkv output features

#if defined(__AMDGCN__) && __has_builtin(__builtin_amdgcn_tensor_load_to_lds) && \
    __has_builtin(__builtin_amdgcn_s_wait_tensorcnt)
#define USE_TDM 1
#else
#define USE_TDM 0
#endif

__device__ inline v8f wmma_f16(v16h a, v16h b, v8f c) {
  // v_wmma_f32_16x16x32_f16: D = A(16x32,f16) * B(32x16,f16) + C(16x16,f32)
  return __builtin_amdgcn_wmma_f32_16x16x32_f16(false, a, false, b,
                                                (short)0, c, false, false);
}

// A/B fragment from f16 memory, element (i, k) = base[i*ld + k] (k contiguous).
// ISA 16-bit A 16x32 layout: lane<16 holds row i=lane with K {0..7}U{16..23};
// lane>=16 same row with K {8..15}U{24..31}.  Each half is one aligned 16B
// chunk -> global_load_b128 / ds_load_b128.
__device__ inline v16h load_fragh_contigK(const _Float16* __restrict__ base,
                                          int ld, int lane) {
  const int i = lane & 15;
  const int khalf = (lane >> 4) << 3;  // 0 or 8
  const _Float16* row = base + (size_t)i * ld + khalf;
  v8h lo = *(const v8h*)(row);         // K khalf .. khalf+7
  v8h hi = *(const v8h*)(row + 16);    // K khalf+16 .. khalf+23
  return __builtin_shufflevector(lo, hi, 0, 1, 2, 3, 4, 5, 6, 7,
                                 8, 9, 10, 11, 12, 13, 14, 15);
}

#if USE_TDM
// Issue one TDM 2D tile load (f16 elements) into LDS.  Descriptor bit layout
// per CDNA5 ISA ch.8 (D# groups): group0 = {count=1, lds_addr, global_addr,
// type=2}; group1 = {data_size=2B, tensor_dim0/1 = tile dims (no OOB),
// tile_dim0/1, tensor_dim0_stride}.  Groups 2/3 zero (2D tile).
__device__ inline void tdm_load_2d(const _Float16* g, unsigned lds_byte,
                                   int tw, int th, int stride_elems) {
  unsigned long long ga = (unsigned long long)(size_t)g;
  tdm_g0_t g0 = {1u,                                       // count=1, user D#
                 lds_byte,                                 // lds_addr
                 (unsigned)ga,                             // global_addr[31:0]
                 (unsigned)((ga >> 32) & 0x01FFFFFFull) | (2u << 30)};
  tdm_g1_t g1 = {(int)0x00010000,                          // data_size=1 (2B)
                 (int)((tw & 0xFFFF) << 16),               // tensor_dim0 lo
                 (int)((((unsigned)tw >> 16) & 0xFFFF) | ((th & 0xFFFF) << 16)),
                 (int)((((unsigned)th >> 16) & 0xFFFF) | ((tw & 0xFFFF) << 16)),
                 (int)(th & 0xFFFF),                       // tile_dim1 (dim2=0)
                 stride_elems,                             // dim0_stride[31:0]
                 0, 0};
  tdm_g2_t z = {0, 0, 0, 0};
#if __clang_major__ >= 23
  tdm_g1_t z8 = {0, 0, 0, 0, 0, 0, 0, 0};
  __builtin_amdgcn_tensor_load_to_lds(g0, g1, z, z, z8, 0);
#else
  __builtin_amdgcn_tensor_load_to_lds(g0, g1, z, z, 0);
#endif
}

__device__ inline unsigned lds_byte_addr(const void* p) {
  // LDS flat aperture: byte offset lives in the low 32 address bits.
  return (unsigned)(unsigned long long)(size_t)p;
}
#endif  // USE_TDM

// ---------------------------------------------------------------------------
// Kernel 0: bulk f32 -> f16 conversion (bandwidth-bound, one pass).
// ---------------------------------------------------------------------------
__global__ void __launch_bounds__(256)
cvt_f32_to_f16_kernel(const float* __restrict__ src, _Float16* __restrict__ dst,
                      int n4) {
  int i = (blockIdx.x * blockDim.x + threadIdx.x);
  if (i < n4) {
    float4 v = ((const float4*)src)[i];
    _Float16* d = dst + (size_t)i * 4;
    d[0] = (_Float16)v.x; d[1] = (_Float16)v.y;
    d[2] = (_Float16)v.z; d[3] = (_Float16)v.w;
  }
}

// ---------------------------------------------------------------------------
// Kernel 1: qkv = x @ w_qkv^T, f16 operands, f32 accumulate.
// q,k planes stored [s][b][h][n][d]; v plane stored TRANSPOSED [b][h][d][n]
// so the attention PV B-fragment is a contiguous-K (b128) load.
// One wave computes a 16x64 strip (4 WMMAs per K-step, A reused 4x).
// ---------------------------------------------------------------------------
__global__ void __launch_bounds__(128)
qkv_gemm_kernel(const _Float16* __restrict__ xh, const _Float16* __restrict__ wh,
                _Float16* __restrict__ qkvh) {
  const int lane = threadIdx.x & 31;
  const int tile = blockIdx.x * 4 + (threadIdx.x >> 5);
  const int mt = tile / (JQKV / 64);   // token tile 0..255
  const int jg = tile % (JQKV / 64);   // feature group 0..47 (64 features)

  const _Float16* arow = xh + (size_t)(mt * 16) * CC;
  const _Float16* brow = wh + (size_t)(jg * 64) * CC;

  v8f acc0 = {}, acc1 = {}, acc2 = {}, acc3 = {};
#pragma unroll 2
  for (int k0 = 0; k0 < CC; k0 += 32) {
    __builtin_prefetch(arow + k0 + 256, 0, 0);
    v16h a = load_fragh_contigK(arow + k0, CC, lane);
    v16h b;
    b = load_fragh_contigK(brow + k0 + (size_t)0  * CC, CC, lane); acc0 = wmma_f16(a, b, acc0);
    b = load_fragh_contigK(brow + k0 + (size_t)16 * CC, CC, lane); acc1 = wmma_f16(a, b, acc1);
    b = load_fragh_contigK(brow + k0 + (size_t)32 * CC, CC, lane); acc2 = wmma_f16(a, b, acc2);
    b = load_fragh_contigK(brow + k0 + (size_t)48 * CC, CC, lane); acc3 = wmma_f16(a, b, acc3);
  }

  const int col = lane & 15;
  const int rowoff = (lane >> 4) << 3;
  const size_t plane = (size_t)NN * DD;
#pragma unroll
  for (int t = 0; t < 4; ++t) {
    const v8f* accp = (t == 0) ? &acc0 : (t == 1) ? &acc1 : (t == 2) ? &acc2 : &acc3;
    int j = jg * 64 + t * 16 + col;
    int s = j >> 10;
    int h = (j >> 6) & 15;
    int d = j & 63;
#pragma unroll
    for (int r = 0; r < 8; ++r) {
      int tok = mt * 16 + rowoff + r;
      int b_ = tok >> 11;
      int n = tok & (NN - 1);
      size_t pbase = (size_t)((s * BB + b_) * HH + h) * plane;
      size_t idx = (s == 2) ? (pbase + (size_t)d * NN + n)    // V transposed
                            : (pbase + (size_t)n * DD + d);   // Q,K row-major
      qkvh[idx] = (_Float16)((*accp)[r]);
    }
  }
}

// ---------------------------------------------------------------------------
// Kernel 2: RoPE in place on q (s=0) and k (s=1) f16 rows, f32 math.
// One thread owns one full 64-element row -> no write hazards.
// ---------------------------------------------------------------------------
__global__ void __launch_bounds__(256)
rope_kernel(_Float16* __restrict__ qk) {
  const int t = blockIdx.x * blockDim.x + threadIdx.x;  // 0 .. 2*B*H*N-1
  const int n = t & (NN - 1);
  _Float16* row = qk + (size_t)t * DD;

  float vals[DD];
#pragma unroll
  for (int i = 0; i < DD; ++i) vals[i] = (float)row[i];

#pragma unroll
  for (int i = 0; i < DD / 2; ++i) {
    // inv_freq = 10000^(-2i/64)
    float inv = __expf(-9.210340371976184f * (float)i * (1.0f / 32.0f));
    float ang = (float)n * inv;
    float sn, cs;
    __sincosf(ang, &sn, &cs);
    float x1 = vals[2 * i];
    float x2 = vals[2 * i + 1];
    row[i]          = (_Float16)(x1 * cs - x2 * sn);
    row[i + DD / 2] = (_Float16)(x2 * cs + x1 * sn);
  }
}

// ---------------------------------------------------------------------------
// Online-softmax update for one 16x32 score strip (two 16x16 WMMA tiles).
// ---------------------------------------------------------------------------
__device__ inline void softmax_update(v8f s0, v8f s1, int lane,
                                      float (&mrow)[8], float (&lrow)[8],
                                      v8f& a0, v8f& a1, v8f& a2, v8f& a3,
                                      _Float16* __restrict__ pp) {
  const float scale = 0.125f;  // 1/sqrt(64)
#pragma unroll
  for (int r = 0; r < 8; ++r) {
    float v0 = s0[r] * scale;
    float v1 = s1[r] * scale;
    float mx = fmaxf(v0, v1);
#pragma unroll
    for (int msk = 1; msk <= 8; msk <<= 1)
      mx = fmaxf(mx, __shfl_xor(mx, msk, 32));
    float mnew = fmaxf(mrow[r], mx);
    float corr = __expf(mrow[r] - mnew);
    float p0 = __expf(v0 - mnew);
    float p1 = __expf(v1 - mnew);
    float rs = p0 + p1;
#pragma unroll
    for (int msk = 1; msk <= 8; msk <<= 1)
      rs += __shfl_xor(rs, msk, 32);
    lrow[r] = lrow[r] * corr + rs;
    mrow[r] = mnew;
    a0[r] *= corr; a1[r] *= corr; a2[r] *= corr; a3[r] *= corr;

    int m = ((lane >> 4) << 3) + r;
    int c = lane & 15;
    pp[m * 32 + c]      = (_Float16)p0;
    pp[m * 32 + 16 + c] = (_Float16)p1;
  }
}

// ---------------------------------------------------------------------------
// Kernel 3: flash attention, all-f16 operands.  One wave per (b,h,32-query
// block); 2 waves/block.  K/V 32-key tiles are staged into LDS by the Tensor
// Data Mover with double buffering (TENSORcnt), so each tile is fetched once
// per wave and feeds 16 WMMAs from ds_load_b128 fragments.
// ---------------------------------------------------------------------------
__global__ void __launch_bounds__(64)
attn_kernel(const _Float16* __restrict__ qkvh, _Float16* __restrict__ attn_out) {
  __shared__ __align__(16) _Float16 kstage[2][2][32 * DD];  // [wave][buf][key][d]
  __shared__ __align__(16) _Float16 vstage[2][2][DD * 32];  // [wave][buf][d][key]
  __shared__ __align__(16) _Float16 pbuf[2][2][16 * 32];

  const int waveInBlk = threadIdx.x >> 5;
  const int lane = threadIdx.x & 31;
  const int wid = blockIdx.x * 2 + waveInBlk;   // 0..2047
  const int qg = wid & 63;                      // 32-row query group
  const int bh = wid >> 6;                      // 0..31
  const int b = bh >> 4;
  const int h = bh & 15;

  const size_t plane = (size_t)NN * DD;
  const _Float16* qbase = qkvh + (size_t)((0 * BB + b) * HH + h) * plane;
  const _Float16* kbase = qkvh + (size_t)((1 * BB + b) * HH + h) * plane;
  const _Float16* vT    = qkvh + (size_t)((2 * BB + b) * HH + h) * plane;  // [d][n]

  // Two query tiles (rows qg*32 .. +31), fragments resident all loop.
  const _Float16* q0 = qbase + (size_t)(qg * 32) * DD;
  const v16h aq00 = load_fragh_contigK(q0, DD, lane);
  const v16h aq01 = load_fragh_contigK(q0 + 32, DD, lane);
  const v16h aq10 = load_fragh_contigK(q0 + 16 * DD, DD, lane);
  const v16h aq11 = load_fragh_contigK(q0 + 16 * DD + 32, DD, lane);

  v8f a00 = {}, a01 = {}, a02 = {}, a03 = {};
  v8f a10 = {}, a11 = {}, a12 = {}, a13 = {};
  float m0[8], l0[8], m1[8], l1[8];
#pragma unroll
  for (int r = 0; r < 8; ++r) {
    m0[r] = -1e30f; l0[r] = 0.0f;
    m1[r] = -1e30f; l1[r] = 0.0f;
  }

  _Float16* pp0 = pbuf[waveInBlk][0];
  _Float16* pp1 = pbuf[waveInBlk][1];

#if USE_TDM
  // Prologue: DMA tile 0 (K: contiguous 32x64; V: strided 64x32, stride N).
  tdm_load_2d(kbase, lds_byte_addr(kstage[waveInBlk][0]), DD, 32, DD);
  tdm_load_2d(vT, lds_byte_addr(vstage[waveInBlk][0]), 32, DD, NN);
#endif

  for (int kt = 0; kt < NN; kt += 32) {
    const int cur = (kt >> 5) & 1;
#if USE_TDM
    if (kt + 32 < NN) {  // issue next tile into alternate buffer, then wait
      const int nxt = cur ^ 1;
      tdm_load_2d(kbase + (size_t)(kt + 32) * DD,
                  lds_byte_addr(kstage[waveInBlk][nxt]), DD, 32, DD);
      tdm_load_2d(vT + (kt + 32),
                  lds_byte_addr(vstage[waveInBlk][nxt]), 32, DD, NN);
      __builtin_amdgcn_s_wait_tensorcnt(2);  // in-order: current tile resident
    } else {
      __builtin_amdgcn_s_wait_tensorcnt(0);
    }
    asm volatile("" ::: "memory");
    const _Float16* ksrc = kstage[waveInBlk][cur];  // [key][d], ld=64
    const _Float16* vsrc = vstage[waveInBlk][cur];  // [d][key], ld=32
    const v16h bk00 = load_fragh_contigK(ksrc + 0,            DD, lane);
    const v16h bk01 = load_fragh_contigK(ksrc + 32,           DD, lane);
    const v16h bk10 = load_fragh_contigK(ksrc + 16 * DD,      DD, lane);
    const v16h bk11 = load_fragh_contigK(ksrc + 16 * DD + 32, DD, lane);
#else
    const v16h bk00 = load_fragh_contigK(kbase + (size_t)kt * DD + 0,  DD, lane);
    const v16h bk01 = load_fragh_contigK(kbase + (size_t)kt * DD + 32, DD, lane);
    const v16h bk10 = load_fragh_contigK(kbase + (size_t)(kt + 16) * DD + 0,  DD, lane);
    const v16h bk11 = load_fragh_contigK(kbase + (size_t)(kt + 16) * DD + 32, DD, lane);
#endif

    // ---- S = Q K^T, both query tiles (K fragments shared) ----
    v8f s0 = {}, s1 = {};
    s0 = wmma_f16(aq00, bk00, s0); s0 = wmma_f16(aq01, bk01, s0);
    s1 = wmma_f16(aq00, bk10, s1); s1 = wmma_f16(aq01, bk11, s1);
    softmax_update(s0, s1, lane, m0, l0, a00, a01, a02, a03, pp0);

    v8f t0 = {}, t1 = {};
    t0 = wmma_f16(aq10, bk00, t0); t0 = wmma_f16(aq11, bk01, t0);
    t1 = wmma_f16(aq10, bk10, t1); t1 = wmma_f16(aq11, bk11, t1);
    softmax_update(t0, t1, lane, m1, l1, a10, a11, a12, a13, pp1);
    __syncthreads();

    // ---- V fragments (shared by both query tiles) ----
#if USE_TDM
    const v16h bv0 = load_fragh_contigK(vsrc + 0 * 16 * 32, 32, lane);
    const v16h bv1 = load_fragh_contigK(vsrc + 1 * 16 * 32, 32, lane);
    const v16h bv2 = load_fragh_contigK(vsrc + 2 * 16 * 32, 32, lane);
    const v16h bv3 = load_fragh_contigK(vsrc + 3 * 16 * 32, 32, lane);
#else
    const v16h bv0 = load_fragh_contigK(vT + (size_t)0  * NN + kt, NN, lane);
    const v16h bv1 = load_fragh_contigK(vT + (size_t)16 * NN + kt, NN, lane);
    const v16h bv2 = load_fragh_contigK(vT + (size_t)32 * NN + kt, NN, lane);
    const v16h bv3 = load_fragh_contigK(vT + (size_t)48 * NN + kt, NN, lane);
#endif

    // ---- O += P V, both query tiles ----
    const v16h ap0 = load_fragh_contigK(pp0, 32, lane);
    a00 = wmma_f16(ap0, bv0, a00); a01 = wmma_f16(ap0, bv1, a01);
    a02 = wmma_f16(ap0, bv2, a02); a03 = wmma_f16(ap0, bv3, a03);

    const v16h ap1 = load_fragh_contigK(pp1, 32, lane);
    a10 = wmma_f16(ap1, bv0, a10); a11 = wmma_f16(ap1, bv1, a11);
    a12 = wmma_f16(ap1, bv2, a12); a13 = wmma_f16(ap1, bv3, a13);
    __syncthreads();
  }

  // ---- epilogue: normalize, write f16 [B, N, C] with C = h*64 + d ----
  const int c = lane & 15;
  const int rowoff = (lane >> 4) << 3;
#pragma unroll
  for (int r = 0; r < 8; ++r) {
    int n = qg * 32 + rowoff + r;
    float inv = 1.0f / l0[r];
    _Float16* o = attn_out + ((size_t)b * NN + n) * CC + h * DD;
    o[0 * 16 + c] = (_Float16)(a00[r] * inv);
    o[1 * 16 + c] = (_Float16)(a01[r] * inv);
    o[2 * 16 + c] = (_Float16)(a02[r] * inv);
    o[3 * 16 + c] = (_Float16)(a03[r] * inv);

    n += 16;
    inv = 1.0f / l1[r];
    o = attn_out + ((size_t)b * NN + n) * CC + h * DD;
    o[0 * 16 + c] = (_Float16)(a10[r] * inv);
    o[1 * 16 + c] = (_Float16)(a11[r] * inv);
    o[2 * 16 + c] = (_Float16)(a12[r] * inv);
    o[3 * 16 + c] = (_Float16)(a13[r] * inv);
  }
}

// ---------------------------------------------------------------------------
// Kernel 4: out = attn_out @ w_proj^T + b_proj, f16 operands, f32 out.
// 16x64 strip per wave.
// ---------------------------------------------------------------------------
__global__ void __launch_bounds__(128)
proj_gemm_kernel(const _Float16* __restrict__ ah, const _Float16* __restrict__ wh,
                 const float* __restrict__ bias, float* __restrict__ out) {
  const int lane = threadIdx.x & 31;
  const int tile = blockIdx.x * 4 + (threadIdx.x >> 5);
  const int mt = tile / (CC / 64);   // token tile 0..255
  const int jg = tile % (CC / 64);   // out-feature group 0..15

  const _Float16* arow = ah + (size_t)(mt * 16) * CC;
  const _Float16* brow = wh + (size_t)(jg * 64) * CC;

  v8f acc0 = {}, acc1 = {}, acc2 = {}, acc3 = {};
#pragma unroll 2
  for (int k0 = 0; k0 < CC; k0 += 32) {
    __builtin_prefetch(brow + k0 + 256, 0, 0);
    v16h a = load_fragh_contigK(arow + k0, CC, lane);
    v16h b;
    b = load_fragh_contigK(brow + k0 + (size_t)0  * CC, CC, lane); acc0 = wmma_f16(a, b, acc0);
    b = load_fragh_contigK(brow + k0 + (size_t)16 * CC, CC, lane); acc1 = wmma_f16(a, b, acc1);
    b = load_fragh_contigK(brow + k0 + (size_t)32 * CC, CC, lane); acc2 = wmma_f16(a, b, acc2);
    b = load_fragh_contigK(brow + k0 + (size_t)48 * CC, CC, lane); acc3 = wmma_f16(a, b, acc3);
  }

  const int col = lane & 15;
  const int rowoff = (lane >> 4) << 3;
#pragma unroll
  for (int t = 0; t < 4; ++t) {
    const v8f* accp = (t == 0) ? &acc0 : (t == 1) ? &acc1 : (t == 2) ? &acc2 : &acc3;
    int j = jg * 64 + t * 16 + col;
    float bj = bias[j];
#pragma unroll
    for (int r = 0; r < 8; ++r) {
      int tok = mt * 16 + rowoff + r;
      out[(size_t)tok * CC + j] = (*accp)[r] + bj;
    }
  }
}

// ---------------------------------------------------------------------------
extern "C" void kernel_launch(void* const* d_in, const int* in_sizes, int n_in,
                              void* d_out, int out_size, void* d_ws,
                              size_t ws_size, hipStream_t stream) {
  const float* x      = (const float*)d_in[0];
  const float* w_qkv  = (const float*)d_in[1];
  const float* w_proj = (const float*)d_in[2];
  const float* b_proj = (const float*)d_in[3];
  float* out = (float*)d_out;

  // f16 workspace layout (48 MB total)
  _Float16* xh    = (_Float16*)d_ws;                     // TOK*CC
  _Float16* wqh   = xh  + (size_t)TOK * CC;              // JQKV*CC
  _Float16* wph   = wqh + (size_t)JQKV * CC;             // CC*CC
  _Float16* qkvh  = wph + (size_t)CC * CC;               // 3*B*H*N*D
  _Float16* attnh = qkvh + (size_t)3 * BB * HH * NN * DD;  // TOK*CC

  // 0) one-pass f32 -> f16 conversion of all GEMM operands
  cvt_f32_to_f16_kernel<<<(TOK * CC / 4) / 256, 256, 0, stream>>>(x, xh, TOK * CC / 4);
  cvt_f32_to_f16_kernel<<<(JQKV * CC / 4) / 256, 256, 0, stream>>>(w_qkv, wqh, JQKV * CC / 4);
  cvt_f32_to_f16_kernel<<<(CC * CC / 4) / 256, 256, 0, stream>>>(w_proj, wph, CC * CC / 4);

  // 1) QKV projection: 256 token tiles x 48 feature groups, 4 waves/block
  qkv_gemm_kernel<<<(TOK / 16) * (JQKV / 64) / 4, 128, 0, stream>>>(xh, wqh, qkvh);

  // 2) RoPE in place on q and k planes: 2*B*H*N rows
  rope_kernel<<<(2 * BB * HH * NN) / 256, 256, 0, stream>>>(qkvh);

  // 3) Flash attention: 32 (b,h) x 64 query groups = 2048 waves, 2/block
  attn_kernel<<<(BB * HH * (NN / 32)) / 2, 64, 0, stream>>>(qkvh, attnh);

  // 4) Output projection + bias
  proj_gemm_kernel<<<(TOK / 16) * (CC / 64) / 4, 128, 0, stream>>>(attnh, wph,
                                                                   b_proj, out);
}
